// SpatialTemporal_MS_GCN_77979426226969
// MI455X (gfx1250) — matrix operand order repaired and checked by reading, not compiled, not run on hardware
//
#include <hip/hip_runtime.h>

// ---------------------------------------------------------------------------
// SpatialTemporal MS-GCN fused block for MI455X (gfx1250, wave32, WMMA).
//
// Both GEMMs run as v_wmma_f32_16x16x32_bf16 with f32 accumulation; A_scale
// and the BN-folded W are pre-converted to bf16 once. Roofline: ~22 GFLOP over
// a minimum of ~370 MB HBM traffic (x read + out write) -> needs ~1.4 PFLOP/s
// to be compute-bound at 23.3 TB/s; only the bf16 matrix pipe delivers that,
// so the kernel is HBM-bound and the 184 MB `agg` intermediate is fused away
// into LDS. Per wave, GEMM B-operands are register-resident across tiles.
// ---------------------------------------------------------------------------

typedef __attribute__((ext_vector_type(16))) __bf16 v16bf;
typedef __attribute__((ext_vector_type(8)))  __bf16 v8bf;
typedef __attribute__((ext_vector_type(2)))  __bf16 v2bf;
typedef __attribute__((ext_vector_type(8)))  float  v8f;

#define Nn   32
#define Cc   96
#define Tt_  300
#define VW   50
#define Vv   25
#define OC   192
#define KC   192   // NUM_SCALES * C

// workspace byte offsets (all 256B-aligned)
#define WS_ANORM 0        // 32*625 f32   = 80000 B
#define WS_APRE  81920    // 64*64 bf16   =  8192 B  (A_scale[v][u], padded, bf16)
#define WS_WBN   90112    // 192*192 bf16 = 73728 B  (BN-folded W, row-major [o][k])
#define WS_BIAS  163840   // 192 f32      (BN-folded bias)

// ------------------------- fragment loaders --------------------------------
// A-matrix 16x32 bf16 fragment (ISA 7.12.2): lane<16 holds K {kb+0..7, kb+16..23}
// of row M=lane%16; lane>=16 holds K {kb+8..15, kb+24..31}. p0 points at the
// first 8 contiguous K values for this lane.
static __device__ __forceinline__ v16bf ldfragA(const __bf16* p0) {
  v8bf lo = *(const v8bf*)(p0);
  v8bf hi = *(const v8bf*)(p0 + 16);
  return __builtin_shufflevector(lo, hi, 0,1,2,3,4,5,6,7,8,9,10,11,12,13,14,15);
}
// B-matrix 32x16 bf16 fragment: column N = lane%16; lane<16 holds K {kb+0..7, kb+16..23},
// lane>=16 holds K {kb+8..15, kb+24..31} (source is [n][k] layout).
static __device__ __forceinline__ v16bf ldfragB(const __bf16* p0) {
  v8bf lo = *(const v8bf*)(p0);
  v8bf hi = *(const v8bf*)(p0 + 16);
  return __builtin_shufflevector(lo, hi, 0,1,2,3,4,5,6,7,8,9,10,11,12,13,14,15);
}

// ---------------------------------------------------------------------------
// Kernel 1: per-sample adjacency A_norm[n] (32 blocks, 256 threads).
// A_t[v][u] = mean_t exp(-2*||xx[n,:,t,v]-xx[n,:,t,u]||^2); D^-1/2 A D^-1/2.
// ---------------------------------------------------------------------------
__global__ __launch_bounds__(256) void k_adj(const float* __restrict__ xx,
                                             float* __restrict__ anorm) {
  __shared__ float fr[3 * Vv];    // one frame, [c][v]
  __shared__ float sA[Vv * Vv];
  __shared__ float sdinv[Vv];
  const int n = blockIdx.x, tid = threadIdx.x;

  float acc0 = 0.f, acc1 = 0.f, acc2 = 0.f;
  for (int t = 0; t < Tt_; ++t) {
    if (tid < 3 * Vv) {
      int c = tid / Vv, v = tid % Vv;
      fr[tid] = xx[((n * 3 + c) * Tt_ + t) * Vv + v];
    }
    __syncthreads();
    #pragma unroll
    for (int i = 0; i < 3; ++i) {
      int p = tid + i * 256;
      if (p < Vv * Vv) {
        int v = p / Vv, u = p % Vv;
        float d0 = fr[v] - fr[u];
        float d1 = fr[Vv + v] - fr[Vv + u];
        float d2 = fr[2 * Vv + v] - fr[2 * Vv + u];
        float s = d0 * d0 + d1 * d1 + d2 * d2;
        float e = __expf(-2.0f * s);          // LAMBDA = 0.5
        if (i == 0) acc0 += e; else if (i == 1) acc1 += e; else acc2 += e;
      }
    }
    __syncthreads();
  }
  #pragma unroll
  for (int i = 0; i < 3; ++i) {
    int p = tid + i * 256;
    if (p < Vv * Vv) sA[p] = (i == 0 ? acc0 : (i == 1 ? acc1 : acc2)) * (1.0f / Tt_);
  }
  __syncthreads();
  if (tid < Vv) {
    float s = 0.f;
    for (int u = 0; u < Vv; ++u) s += sA[tid * Vv + u];
    sdinv[tid] = rsqrtf(s);
  }
  __syncthreads();
  #pragma unroll
  for (int i = 0; i < 3; ++i) {
    int p = tid + i * 256;
    if (p < Vv * Vv) {
      int v = p / Vv, u = p % Vv;
      anorm[n * (Vv * Vv) + p] = sA[p] * sdinv[v] * sdinv[u];
    }
  }
}

// ---------------------------------------------------------------------------
// Kernel 2 (1 block): batch-mean + tile + A_res -> bf16 A_scale (64x64 pad);
// fold BN into W (bf16) and bias (f32).
// ---------------------------------------------------------------------------
__global__ __launch_bounds__(256) void k_prep(const float* __restrict__ anorm,
                                              const float* __restrict__ a_res,
                                              const float* __restrict__ W,
                                              const float* __restrict__ b,
                                              const float* __restrict__ gamma,
                                              const float* __restrict__ beta,
                                              const float* __restrict__ rmean,
                                              const float* __restrict__ rvar,
                                              __bf16* __restrict__ apre,
                                              __bf16* __restrict__ wbn,
                                              float* __restrict__ biasbn) {
  __shared__ float sAs[Vv * Vv];
  __shared__ float sSc[OC];
  const int tid = threadIdx.x;
  for (int e = tid; e < Vv * Vv; e += 256) {
    float s = 0.f;
    for (int n = 0; n < Nn; ++n) s += anorm[n * (Vv * Vv) + e];
    sAs[e] = s * (1.0f / Nn);
  }
  for (int o = tid; o < OC; o += 256) {
    float sc = gamma[o] * rsqrtf(rvar[o] + 1e-5f);
    sSc[o] = sc;
    biasbn[o] = beta[o] + (b[o] - rmean[o]) * sc;
  }
  __syncthreads();
  for (int idx = tid; idx < 64 * 64; idx += 256) {
    int v = idx >> 6, u = idx & 63;
    float val = 0.f;
    if (v < VW && u < VW) val = sAs[(v % Vv) * Vv + (u % Vv)] + a_res[v * VW + u];
    apre[idx] = (__bf16)val;                  // [v][u] layout == B-matrix [n][k]
  }
  for (int idx = tid; idx < OC * KC; idx += 256) {
    int o = idx / KC;
    wbn[idx] = (__bf16)(W[idx] * sSc[o]);     // row-major [o][k]
  }
}

// ---------------------------------------------------------------------------
// Kernel 3: fused  agg = x·A^T  ->  scale-shuffle  ->  out = Wbn·agg + bias,
// ReLU. Block = (n, 2 timesteps). grid (150, 32), 256 threads = 8 waves.
// Wave partitioning pins the GEMM-B operand per wave so B-fragments stay in
// registers across all output tiles.
// ---------------------------------------------------------------------------
__global__ __launch_bounds__(256) void k_gcn(const float* __restrict__ x,
                                             const __bf16* __restrict__ apre,
                                             const __bf16* __restrict__ wbn,
                                             const float* __restrict__ biasbn,
                                             float* __restrict__ out) {
  __shared__ __bf16 sX[192 * 64];     // rows (c*2+tt), K=u padded to 64
  __shared__ __bf16 sAggT[64 * 192];  // [j][k]: j = tt*25+vs (pad 64), k = s*96+c
  const int tc = blockIdx.x;          // 0..149 (time chunk of 2)
  const int n  = blockIdx.y;          // 0..31
  const int tid = threadIdx.x;
  const int wid = tid >> 5, lane = tid & 31;
  const int hl = lane >> 4, lr = lane & 15;

  // stage x -> bf16 LDS (zero-padded K); paired loads + packed bf16 converts
  for (int e = tid; e < 192 * 32; e += 256) {
    int row = e >> 5, u = (e & 31) * 2;
    int c = row >> 1, tt = row & 1;
    float xa = 0.f, xb = 0.f;
    if (u < VW) {
      float2 xv = *(const float2*)(x + ((n * Cc + c) * Tt_ + (tc * 2 + tt)) * VW + u);
      xa = xv.x; xb = xv.y;
    }
    v2bf pk = {(__bf16)xa, (__bf16)xb};
    *(v2bf*)(sX + row * 64 + u) = pk;
  }
  for (int e = tid; e < 14 * KC; e += 256) sAggT[VW * KC + e] = (__bf16)0.f;
  __syncthreads();

  // ---- G1: agg[(c,tt), v] = sum_u x * A[v][u] --------------------------
  // wave pins ntile (nt = wid&3); the two A_scale B-frags live in registers.
  {
    const int nt = wid & 3;
    const v16bf b0 = ldfragB(apre + (nt * 16 + lr) * 64 + 0  + hl * 8);
    const v16bf b1 = ldfragB(apre + (nt * 16 + lr) * 64 + 32 + hl * 8);
    for (int mt = (wid >> 2); mt < 12; mt += 2) {
      v16bf a0 = ldfragA(sX + (mt * 16 + lr) * 64 + 0  + hl * 8);
      v16bf a1 = ldfragA(sX + (mt * 16 + lr) * 64 + 32 + hl * 8);
      v8f acc = {0.f,0.f,0.f,0.f,0.f,0.f,0.f,0.f};
      acc = __builtin_amdgcn_wmma_f32_16x16x32_bf16(false, a0, false, b0,
                                                    (short)0, acc, false, false);
      acc = __builtin_amdgcn_wmma_f32_16x16x32_bf16(false, a1, false, b1,
                                                    (short)0, acc, false, false);
      #pragma unroll
      for (int r = 0; r < 8; ++r) {         // D layout: M = r + 8*hl, N = lr
        int m1 = mt * 16 + r + hl * 8;
        int v  = nt * 16 + lr;
        if (v < VW) {
          int c = m1 >> 1, tt = m1 & 1;
          int s = (v >= Vv) ? 1 : 0;
          sAggT[(tt * Vv + (v - s * Vv)) * KC + (s * Cc + c)] = (__bf16)acc[r];
        }
      }
    }
  }
  __syncthreads();

  // ---- G2: out[o, j] = sum_k Wbn[o][k] * aggT[j][k] --------------------
  // wave pins jtile (jt = wid&3); all six sAggT B-frags live in registers,
  // Wbn A-frags stream from L2.
  {
    const int jt = wid & 3;
    v16bf bf[6];
    #pragma unroll
    for (int ks = 0; ks < 6; ++ks)
      bf[ks] = ldfragB(sAggT + (jt * 16 + lr) * KC + ks * 32 + hl * 8);
    for (int ot = (wid >> 2); ot < 12; ot += 2) {
      v8f acc = {0.f,0.f,0.f,0.f,0.f,0.f,0.f,0.f};
      #pragma unroll
      for (int ks = 0; ks < 6; ++ks) {
        v16bf af = ldfragA(wbn + (ot * 16 + lr) * KC + ks * 32 + hl * 8);
        acc = __builtin_amdgcn_wmma_f32_16x16x32_bf16(false, af, false, bf[ks],
                                                      (short)0, acc, false, false);
      }
      #pragma unroll
      for (int r = 0; r < 8; ++r) {
        int o = ot * 16 + r + hl * 8;
        int j = jt * 16 + lr;
        if (j < 2 * Vv) {
          int tt = (j >= Vv) ? 1 : 0;
          int vs = j - tt * Vv;
          float val = acc[r] + biasbn[o];
          out[((n * OC + o) * Tt_ + tc * 2 + tt) * Vv + vs] = fmaxf(val, 0.f);
        }
      }
    }
  }
}

// ---------------------------------------------------------------------------
extern "C" void kernel_launch(void* const* d_in, const int* in_sizes, int n_in,
                              void* d_out, int out_size, void* d_ws, size_t ws_size,
                              hipStream_t stream) {
  const float* x     = (const float*)d_in[0];
  const float* xx    = (const float*)d_in[1];
  const float* a_res = (const float*)d_in[2];
  const float* W     = (const float*)d_in[3];
  const float* b     = (const float*)d_in[4];
  const float* gamma = (const float*)d_in[5];
  const float* beta  = (const float*)d_in[6];
  const float* rmean = (const float*)d_in[7];
  const float* rvar  = (const float*)d_in[8];

  float*  wsAnorm = (float*)((char*)d_ws + WS_ANORM);
  __bf16* wsApre  = (__bf16*)((char*)d_ws + WS_APRE);
  __bf16* wsWbn   = (__bf16*)((char*)d_ws + WS_WBN);
  float*  wsBias  = (float*)((char*)d_ws + WS_BIAS);

  k_adj<<<Nn, 256, 0, stream>>>(xx, wsAnorm);
  k_prep<<<1, 256, 0, stream>>>(wsAnorm, a_res, W, b, gamma, beta, rmean, rvar,
                                wsApre, wsWbn, wsBias);
  k_gcn<<<dim3(Tt_ / 2, Nn), 256, 0, stream>>>(x, wsApre, wsWbn, wsBias,
                                               (float*)d_out);
}
